// VectorQuantizer_49435073577316
// MI455X (gfx1250) — matrix operand order repaired and checked by reading, not compiled, not run on hardware
//
#include <hip/hip_runtime.h>
#include <stdint.h>

typedef __attribute__((ext_vector_type(16))) _Float16 v16h;
typedef __attribute__((ext_vector_type(8)))  _Float16 v8h;
typedef __attribute__((ext_vector_type(4)))  _Float16 v4h;
typedef __attribute__((ext_vector_type(8)))  float    v8f;

#define NPTS 65536   // 64*32*32 points
#define DIM  256     // embedding dim
#define KCB  1024    // num codebook entries
#define HW   1024    // 32*32
#define NB   64      // batch
#define BROW 264     // padded LDS row stride (halfs): 528B -> 16-lane b128 reads rotate all 64 banks

// Use CDNA5 async global->LDS copies (ASYNCcnt) for codebook tile staging.
#define VQ_ASYNC 1

// ---------------- zero scalar accumulator ----------------
__global__ void vq_zero(float* p) {
    if (threadIdx.x == 0) *p = 0.0f;
}

// ---------------- codebook prep: f32 -> f16 + ||c||^2 ----------------
__global__ __launch_bounds__(256) void vq_prep_codebook(
    const float* __restrict__ cb, _Float16* __restrict__ cb16,
    float* __restrict__ cnorm) {
    int k = blockIdx.x;
    int t = threadIdx.x;
    float v = cb[(size_t)k * DIM + t];
    cb16[(size_t)k * DIM + t] = (_Float16)v;
    __shared__ float red[256];
    red[t] = v * v;
    __syncthreads();
    for (int s = 128; s > 0; s >>= 1) {
        if (t < s) red[t] += red[t + s];
        __syncthreads();
    }
    if (t == 0) cnorm[k] = red[0];
}

// ---------------- x: [B, D, H, W] f32 -> [N, D] f16 (LDS-tiled transpose) ----------------
__global__ __launch_bounds__(256) void vq_transpose_x(
    const float* __restrict__ x, _Float16* __restrict__ xf16) {
    __shared__ float tile[32][33];
    int b  = blockIdx.z;
    int d0 = blockIdx.y * 32;
    int h0 = blockIdx.x * 32;   // hw offset
    int tx = threadIdx.x, ty = threadIdx.y;
    const float* xb = x + (size_t)b * DIM * HW;
#pragma unroll
    for (int i = 0; i < 4; i++) {
        int d = d0 + ty + i * 8;
        tile[ty + i * 8][tx] = xb[(size_t)d * HW + h0 + tx];   // coalesced over hw
    }
    __syncthreads();
    // Packed write: 256 threads = 32 hw-rows x 8 d-segments of 4 halfs -> b64 stores.
    int t  = ty * 32 + tx;
    int r  = t >> 3;            // hw row within tile (0..31)
    int sg = t & 7;             // d segment (0..7), 4 halfs each
    int n  = b * HW + h0 + r;
    v4h val;
#pragma unroll
    for (int e = 0; e < 4; e++) val[e] = (_Float16)tile[sg * 4 + e][r];
    *(v4h*)(xf16 + (size_t)n * DIM + d0 + sg * 4) = val;
}

// ---------------- async tile staging helpers ----------------
__device__ __forceinline__ void vq_stage_tile(const _Float16* __restrict__ cb16,
                                              _Float16* lbase, int nc, int t) {
    int r   = t >> 4;   // code row within tile (0..15)
    int seg = t & 15;   // 32B segment (0..15)
    const _Float16* g = cb16 + (size_t)(nc * 16 + r) * DIM + seg * 16;
    _Float16* l = lbase + r * BROW + seg * 16;
#if VQ_ASYNC
    uint32_t lds32 = (uint32_t)(uintptr_t)l;      // LDS aperture: low 32 bits = LDS byte offset
    uint64_t g64   = (uint64_t)(uintptr_t)g;
    asm volatile("global_load_async_to_lds_b128 %0, %1, off\n\t"
                 "global_load_async_to_lds_b128 %0, %1, off offset:16"
                 :: "v"(lds32), "v"(g64) : "memory");
#else
    v8h a = *(const v8h*)g;
    v8h b = *(const v8h*)(g + 8);
    *(v8h*)l       = a;
    *(v8h*)(l + 8) = b;
#endif
}

__device__ __forceinline__ void vq_wait_async_le2() {
#if VQ_ASYNC
    asm volatile("s_wait_asynccnt 0x2" ::: "memory");
#endif
}
__device__ __forceinline__ void vq_wait_async_le0() {
#if VQ_ASYNC
    asm volatile("s_wait_asynccnt 0x0" ::: "memory");
#endif
}

// ---------------- WMMA distance GEMM + argmin ----------------
// score_k = ||c_k||^2 - 2 * x.c_k   (||x||^2 constant per point -> irrelevant for argmin)
// Block = 256 threads = 8 waves; wave handles 16 points x all 1024 codes.
// Codebook tiles (16 codes x 256 d, 8KB) staged into LDS once per block,
// double-buffered via async global->LDS loads; WMMAs consume via ds_load_b128.
__global__ __launch_bounds__(256) void vq_argmin(
    const _Float16* __restrict__ xf16, const _Float16* __restrict__ cb16,
    const float* __restrict__ cnorm, int* __restrict__ idxout) {
    __shared__ alignas(16) _Float16 ldsB[2][16 * BROW];

    int tid  = threadIdx.x;
    int wave = tid >> 5;
    int lane = tid & 31;
    int lh   = lane & 15;   // A row / B column within 16x16 tile
    int hf   = lane >> 4;   // lane half selects K sub-range
    int m0   = blockIdx.x * 128 + wave * 16;

    // Hoist A fragments (16 points x 256 dim = 8 k-chunks of 32).
    // Dense-A f16 layout: lanes 0-15 row M=lh hold K {0..7,16..23}; lanes 16-31 hold {8..15,24..31}.
    const _Float16* xrow = xf16 + (size_t)(m0 + lh) * DIM;
    v16h afrag[8];
#pragma unroll
    for (int kc = 0; kc < 8; kc++) {
        v8h lo = *(const v8h*)(xrow + kc * 32 + hf * 8);
        v8h hi = *(const v8h*)(xrow + kc * 32 + 16 + hf * 8);
#pragma unroll
        for (int e = 0; e < 8; e++) { afrag[kc][e] = lo[e]; afrag[kc][e + 8] = hi[e]; }
    }

    float best[8];
    int   bidx[8];
#pragma unroll
    for (int j = 0; j < 8; j++) { best[j] = 3.0e38f; bidx[j] = 0; }

    vq_stage_tile(cb16, &ldsB[0][0], 0, tid);   // prologue: tile 0 -> buf 0

    for (int nc = 0; nc < 64; ++nc) {
        int cur = nc & 1;
        if (nc + 1 < 64) {
            vq_stage_tile(cb16, &ldsB[cur ^ 1][0], nc + 1, tid);  // prefetch next tile
            vq_wait_async_le2();                                  // current tile resident
        } else {
            vq_wait_async_le0();
        }
        __syncthreads();   // staged writes visible to all waves

        int n = nc * 16 + lh;           // code index for this lane's column
        const _Float16* lb = &ldsB[cur][lh * BROW];
        v8f acc = {};
#pragma unroll
        for (int kc = 0; kc < 8; kc++) {
            // B f16 layout: lanes 0-15 hold K {0..7,16..23}, lanes 16-31 hold K {8..15,24..31}
            const _Float16* bp = lb + kc * 32 + hf * 8;
            v8h lo = *(const v8h*)(bp);
            v8h hi = *(const v8h*)(bp + 16);
            v16h bfrag;
#pragma unroll
            for (int e = 0; e < 8; e++) { bfrag[e] = lo[e]; bfrag[e + 8] = hi[e]; }
            acc = __builtin_amdgcn_wmma_f32_16x16x32_f16(
                false, afrag[kc], false, bfrag, (short)0, acc, false, false);
        }
        float cn = cnorm[n];
#pragma unroll
        for (int j = 0; j < 8; j++) {
            float s = __builtin_fmaf(-2.0f, acc[j], cn);
            if (s < best[j]) { best[j] = s; bidx[j] = n; }   // strict < keeps first index per lane
        }
        __syncthreads();   // all waves done with buf[cur] before it is restaged
    }

    // Reduce across the 16 lanes holding the same row (C layout: N = lane&15, M = j + 8*hf).
#pragma unroll
    for (int j = 0; j < 8; j++) {
        float bs = best[j];
        int   bi = bidx[j];
#pragma unroll
        for (int off = 8; off >= 1; off >>= 1) {
            float os = __shfl_xor(bs, off, 32);
            int   oi = __shfl_xor(bi, off, 32);
            if (os < bs || (os == bs && oi < bi)) { bs = os; bi = oi; }  // first argmin
        }
        if (lh == 0) idxout[m0 + hf * 8 + j] = bi;
    }
}

// ---------------- gather q (coalesced in [B,D,H,W] order) + MSE accumulation ----------------
__global__ __launch_bounds__(256) void vq_gather_loss(
    const float* __restrict__ x, const float* __restrict__ cb,
    const int* __restrict__ idx, float* __restrict__ out,
    float* __restrict__ sumsq) {
    int t = threadIdx.x;
    size_t e = (size_t)blockIdx.x * 256 + t;      // flat index into [B, D, H, W]
    int b  = (int)(e >> 18);
    int d  = (int)((e >> 10) & 255);
    int hw = (int)(e & 1023);
    int n  = b * HW + hw;
    float q = cb[(size_t)idx[n] * DIM + d];
    out[e] = q;                                   // straight-through output == q
    float diff = x[e] - q;
    __shared__ float red[256];
    red[t] = diff * diff;
    __syncthreads();
    for (int s = 128; s > 0; s >>= 1) {
        if (t < s) red[t] += red[t + s];
        __syncthreads();
    }
    if (t == 0) atomicAdd(sumsq, red[0]);
}

// ---------------- finalize: vq_loss = 1.25 * mean((x-q)^2) ----------------
__global__ void vq_finalize(const float* __restrict__ sumsq, float* __restrict__ outloss) {
    if (threadIdx.x == 0) *outloss = 1.25f * (*sumsq) / 16777216.0f;
}

extern "C" void kernel_launch(void* const* d_in, const int* in_sizes, int n_in,
                              void* d_out, int out_size, void* d_ws, size_t ws_size,
                              hipStream_t stream) {
    const float* x  = (const float*)d_in[0];   // [64, 256, 32, 32]
    const float* cb = (const float*)d_in[1];   // [1024, 256]
    float* out = (float*)d_out;                // quantized (16777216) + vq_loss (1)

    // workspace layout
    char* ws = (char*)d_ws;
    _Float16* xf16  = (_Float16*)(ws);                                   // 32 MB
    _Float16* cb16  = (_Float16*)(ws + (size_t)NPTS * DIM * 2);          // 512 KB
    float*    cnorm = (float*)(ws + (size_t)NPTS * DIM * 2 + (size_t)KCB * DIM * 2);
    int*      idx   = (int*)((char*)cnorm + KCB * sizeof(float));
    float*    sumsq = (float*)((char*)idx + NPTS * sizeof(int));

    vq_zero<<<1, 64, 0, stream>>>(sumsq);
    vq_prep_codebook<<<KCB, 256, 0, stream>>>(cb, cb16, cnorm);
    vq_transpose_x<<<dim3(HW / 32, DIM / 32, NB), dim3(32, 8, 1), 0, stream>>>(x, xf16);
    vq_argmin<<<NPTS / 128, 256, 0, stream>>>(xf16, cb16, cnorm, idx);
    vq_gather_loss<<<(NPTS * DIM) / 256, 256, 0, stream>>>(x, cb, idx, out, sumsq);
    vq_finalize<<<1, 64, 0, stream>>>(sumsq, out + (out_size - 1));
}
